// Q_ResMLP24_55800215110166
// MI455X (gfx1250) — compile-verified, matrix operand order checked by reading, not run on hardware
//
#include <hip/hip_runtime.h>
#include <stdint.h>

// ---------------------------------------------------------------------------
// Quantized ResMLP-24 forward for MI455X (gfx1250, wave32, WMMA IU8 path)
// Round 4: incremental pointer arithmetic + rotating buffer indices (no %3
// division, no in-loop branch), peeled last iteration. Async global->LDS
// staging (ASYNCcnt) with triple-buffered LDS, one barrier per K-step.
// ---------------------------------------------------------------------------

typedef __attribute__((ext_vector_type(8))) int v8i;
typedef __attribute__((ext_vector_type(4))) int v4i;

#define Bb   64
#define Nn   196
#define Dd   384
#define HIDD 1536
#define NBLK 24
#define M_TOK (Bb * Nn)   // 12544  (= 98 * 128)
#define M_CH  (Bb * Dd)   // 24576  (= 192 * 128)

#if defined(__AMDGCN__) && __has_builtin(__builtin_amdgcn_global_load_async_to_lds_b128)
#define ASYNC_LDS 1
typedef __attribute__((address_space(1))) v4i gv4i_as;  // global v4i
typedef __attribute__((address_space(3))) v4i lv4i_as;  // LDS v4i
#else
#define ASYNC_LDS 0
#endif

// 16-byte global->LDS copy: async DMA when available, else via VGPRs.
__device__ __forceinline__ void cp16B(const int8_t* g, int8_t* l) {
#if ASYNC_LDS
  __builtin_amdgcn_global_load_async_to_lds_b128((gv4i_as*)g, (lv4i_as*)l, 0, 0);
#else
  *(int4*)l = *(const int4*)g;
#endif
}

template <int N>
__device__ __forceinline__ void wait_async() {
#if ASYNC_LDS
#if __has_builtin(__builtin_amdgcn_s_wait_asynccnt)
  __builtin_amdgcn_s_wait_asynccnt(N);
#else
  asm volatile("s_wait_asynccnt 0x0" ::: "memory");
#endif
#endif
}

// ------------------------------- prep kernels ------------------------------

__global__ void k_absmax(const float* __restrict__ w, int n, unsigned* __restrict__ slot) {
  __shared__ unsigned red[256];
  unsigned m = 0u;
  for (int i = blockIdx.x * 256 + threadIdx.x; i < n; i += gridDim.x * 256)
    m = max(m, __float_as_uint(fabsf(w[i])));
  red[threadIdx.x] = m;
  __syncthreads();
  for (int s = 128; s > 0; s >>= 1) {
    if ((int)threadIdx.x < s) red[threadIdx.x] = max(red[threadIdx.x], red[threadIdx.x + s]);
    __syncthreads();
  }
  if (threadIdx.x == 0) atomicMax(slot, red[0]);
}

__global__ void k_wquant(const float* __restrict__ W, int8_t* __restrict__ Wq,
                         int rows, int cols, int cols_pad, int total_pad,
                         const unsigned* __restrict__ slot) {
  const float ws  = __uint_as_float(*slot) * (1.0f / 127.0f);
  const float inv = 1.0f / ws;
  for (int i = blockIdx.x * 256 + threadIdx.x; i < total_pad; i += gridDim.x * 256) {
    int r = i / cols_pad, c = i % cols_pad;
    int8_t q = 0;
    if (r < rows && c < cols) {
      float v = rintf(W[(size_t)r * cols + c] * inv);
      v = fminf(fmaxf(v, -128.0f), 127.0f);
      q = (int8_t)(int)v;
    }
    Wq[i] = q;
  }
}

__global__ void k_bquant(const float* __restrict__ b, int* __restrict__ bq,
                         int n, int n_pad, const unsigned* __restrict__ slot,
                         const float* __restrict__ a_s_p) {
  int i = blockIdx.x * 256 + threadIdx.x;
  if (i >= n_pad) return;
  const float ws = __uint_as_float(*slot) * (1.0f / 127.0f);
  const float as = *a_s_p;
  bq[i] = (i < n) ? (int)rintf(b[i] / (as * ws)) : 0;
}

// im2col + input quantization: fp32 [B,3,224,224] -> int8 [B*196][768]
__global__ void k_im2col(const float* __restrict__ x, int8_t* __restrict__ out,
                         const float* __restrict__ s_in_p) {
  int i = blockIdx.x * 256 + threadIdx.x;
  const int total = M_TOK * 768;
  if (i >= total) return;
  const float inv = 1.0f / *s_in_p;
  int k = i % 768, m = i / 768;
  int b = m / Nn, p = m % Nn;
  int c = k >> 8, rr = k & 255, ky = rr >> 4, kx = rr & 15;
  int py = p / 14, px = p % 14;
  float v = x[(((size_t)b * 3 + c) * 224 + (py * 16 + ky)) * 224 + (px * 16 + kx)];
  float q = fminf(fmaxf(rintf(v * inv), -128.0f), 127.0f);
  out[i] = (int8_t)(int)q;
}

// ------------------------------ WMMA fragments -----------------------------
// Tile in LDS: 16 rows x 64 bytes (row-major K). CDNA5 8-bit A 16x64 layout:
// lane L -> row = L&15, K byte groups {0..7|16..23|32..39|48..55} + (L>=16?8:0).
// B (64x16) mirrors this with lane -> column when the weight tile is stored as
// 16 N-rows x 64 K-bytes.
__device__ __forceinline__ v8i frag_ld(const int8_t* tile) {
  const int lane = threadIdx.x & 31;
  const int8_t* p = tile + (lane & 15) * 64 + ((lane >> 4) << 3);
  const int2 t0 = *(const int2*)(p);
  const int2 t1 = *(const int2*)(p + 16);
  const int2 t2 = *(const int2*)(p + 32);
  const int2 t3 = *(const int2*)(p + 48);
  v8i f;
  f[0] = t0.x; f[1] = t0.y; f[2] = t1.x; f[3] = t1.y;
  f[4] = t2.x; f[5] = t2.y; f[6] = t3.x; f[7] = t3.y;
  return f;
}

// int16 A tile kept raw in LDS (16 rows x 64 shorts = 128B rows); split into
// unsigned-lo / signed-hi byte fragments here (VALU co-executes with WMMA).
__device__ __forceinline__ void frag_ld16(const int8_t* tileB, v8i* lo, v8i* hi) {
  const int lane = threadIdx.x & 31;
  const int8_t* p = tileB + (lane & 15) * 128 + ((lane >> 4) << 4);
  v8i l, h;
#pragma unroll
  for (int g = 0; g < 4; ++g) {
    const int4 t = *(const int4*)(p + g * 32);
    auto lo16 = [](int d) { return (d & 0xFF) | ((d >> 8) & 0xFF00); };
    auto hi16 = [](int d) { return ((d >> 8) & 0xFF) | ((d >> 16) & 0xFF00); };
    l[2 * g]     = lo16(t.x) | (lo16(t.y) << 16);
    l[2 * g + 1] = lo16(t.z) | (lo16(t.w) << 16);
    h[2 * g]     = hi16(t.x) | (hi16(t.y) << 16);
    h[2 * g + 1] = hi16(t.z) | (hi16(t.w) << 16);
  }
  *lo = l;
  *hi = h;
}

// ------------------------------- GEMM kernel -------------------------------
enum {
  MODE_ACT8 = 0,      // int8 out [m*stride + n]
  MODE_ACT8_RELU,     // + relu clip
  MODE_TRANS_T8,      // norm1 out -> T8[b][d][n] stride 256 (K-padded)
  MODE_FROM_T8,       // attn out  -> X[(b*196+n)*384 + d], guard n<196
  MODE_RES16,         // patch embed -> residual int16
  MODE_RESADD16,      // residual add + requant16 (in place on res16)
  MODE_RESADD_F32     // last block: fp32 mixed output
};

template <bool A16>
__global__ __launch_bounds__(256) void k_gemm(
    const void* __restrict__ Aptr, int lda,
    const int8_t* __restrict__ Wq, int ldw,
    const int* __restrict__ bq, int K, int NoutReal,
    int mode, void* __restrict__ outp, int out_stride,
    const unsigned* __restrict__ ws_slot,
    const float* __restrict__ a_s_p,
    const float* __restrict__ s_out_p,
    const float* __restrict__ res_s_p,
    int16_t* __restrict__ res16) {
  constexpr int ATILE = A16 ? 128 * 128 : 128 * 64;  // bytes per A buffer
  constexpr int BTILE = 64 * 64;
  constexpr int LPT   = A16 ? 5 : 3;                 // async loads per thread/buffer
  __shared__ alignas(16) int8_t As[3][ATILE];
  __shared__ alignas(16) int8_t Bs[3][BTILE];

  const int tid  = threadIdx.x;
  const int wave = tid >> 5;
  const int wm   = wave >> 1;  // 0..3 -> 32-row slice
  const int wn   = wave & 1;   // 0..1 -> 32-col slice
  const int m0   = blockIdx.x * 128;
  const int n0   = blockIdx.y * 64;

  const int8_t* A8   = (const int8_t*)Aptr;  // bytes (lda in elements)
  int8_t* lAs        = &As[0][0];
  int8_t* lBs        = &Bs[0][0];

  // ------- per-thread staging addresses (advanced incrementally) -------
  // B tile: 64 N-rows x 64B, one 16B chunk per thread.
  const int8_t* gB = Wq + (size_t)(n0 + (tid >> 2)) * ldw + ((tid & 3) << 4);
  const int lB     = (tid >> 2) * 64 + ((tid & 3) << 4);
  // A tile chunks.
  const int8_t* gA0 = nullptr;
  const int8_t* gA1 = nullptr;
  const int8_t* gA2 = nullptr;
  const int8_t* gA3 = nullptr;
  int lA0 = 0, lA1 = 0, lA2 = 0, lA3 = 0;
  if constexpr (!A16) {
    const int row = tid >> 2, col = (tid & 3) << 4;
    gA0 = A8 + (size_t)(m0 + row) * lda + col;
    gA1 = gA0 + (size_t)64 * lda;
    lA0 = row * 64 + col;
    lA1 = lA0 + 64 * 64;
  } else {
    const int row = tid >> 3, colB = (tid & 7) << 4;  // 128B rows of int16
    gA0 = A8 + ((size_t)(m0 + row) * lda) * 2 + colB;
    gA1 = gA0 + (size_t)32 * lda * 2;
    gA2 = gA0 + (size_t)64 * lda * 2;
    gA3 = gA0 + (size_t)96 * lda * 2;
    lA0 = row * 128 + colB;
    lA1 = lA0 + 32 * 128;
    lA2 = lA0 + 64 * 128;
    lA3 = lA0 + 96 * 128;
  }

  v8i accLo[2][2];
  v8i accHi[2][2];
#pragma unroll
  for (int a = 0; a < 2; ++a)
#pragma unroll
    for (int b = 0; b < 2; ++b) { accLo[a][b] = (v8i)(0); accHi[a][b] = (v8i)(0); }

  auto issue = [&](int buf) {
    int8_t* dstA = lAs + buf * ATILE;
    int8_t* dstB = lBs + buf * BTILE;
    if constexpr (!A16) {
      cp16B(gA0, dstA + lA0);
      cp16B(gA1, dstA + lA1);
      gA0 += 64;
      gA1 += 64;
    } else {
      cp16B(gA0, dstA + lA0);
      cp16B(gA1, dstA + lA1);
      cp16B(gA2, dstA + lA2);
      cp16B(gA3, dstA + lA3);
      gA0 += 128;
      gA1 += 128;
      gA2 += 128;
      gA3 += 128;
    }
    cp16B(gB, dstB + lB);
    gB += 64;
  };

  auto compute = [&](int buf) {
    const int8_t* as0 = lAs + buf * ATILE;
    const int8_t* bs0 = lBs + buf * BTILE;
    v8i bfr[2];
    bfr[0] = frag_ld(bs0 + (wn * 32 + 0) * 64);
    bfr[1] = frag_ld(bs0 + (wn * 32 + 16) * 64);
    if constexpr (!A16) {
      v8i alo[2];
      alo[0] = frag_ld(as0 + (wm * 32 + 0) * 64);
      alo[1] = frag_ld(as0 + (wm * 32 + 16) * 64);
#pragma unroll
      for (int im = 0; im < 2; ++im)
#pragma unroll
        for (int in = 0; in < 2; ++in)
          accLo[im][in] = __builtin_amdgcn_wmma_i32_16x16x64_iu8(
              true, alo[im], true, bfr[in], accLo[im][in], false, false);
    } else {
      v8i alo[2], ahi[2];
      frag_ld16(as0 + (wm * 32 + 0) * 128, &alo[0], &ahi[0]);
      frag_ld16(as0 + (wm * 32 + 16) * 128, &alo[1], &ahi[1]);
#pragma unroll
      for (int im = 0; im < 2; ++im)
#pragma unroll
        for (int in = 0; in < 2; ++in) {
          accLo[im][in] = __builtin_amdgcn_wmma_i32_16x16x64_iu8(
              false, alo[im], true, bfr[in], accLo[im][in], false, false); // lo unsigned
          accHi[im][in] = __builtin_amdgcn_wmma_i32_16x16x64_iu8(
              true, ahi[im], true, bfr[in], accHi[im][in], false, false);  // hi signed
        }
    }
  };

  const int kSteps = K >> 6;
  issue(0);
  int bufc = 0, bufn = 1;
  for (int kt = 0; kt < kSteps - 1; ++kt) {
    issue(bufn);
    wait_async<LPT>();  // our loads for buf `bufc` are older -> complete
    __syncthreads();    // everyone's loads for `bufc` done; prior reads done
    compute(bufc);
    bufc = bufn;
    bufn = (bufn == 2) ? 0 : bufn + 1;
  }
  wait_async<0>();
  __syncthreads();
  compute(bufc);

  // ------------------------------ epilogue ------------------------------
  const float ws = __uint_as_float(*ws_slot) * (1.0f / 127.0f);
  const float as = *a_s_p;
  float mul = as * ws;
  float m2  = 0.0f;
  if (mode == MODE_RESADD16 || mode == MODE_RESADD_F32) m2 = *res_s_p;
  if (mode != MODE_RESADD_F32) {
    const float so = *s_out_p;
    mul /= so;
    m2  /= so;
  }

  const int lane    = threadIdx.x & 31;
  const int rowBase = (lane >> 4) << 3;
  const int colIn   = lane & 15;
#pragma unroll
  for (int im = 0; im < 2; ++im)
#pragma unroll
    for (int in = 0; in < 2; ++in)
#pragma unroll
      for (int r = 0; r < 8; ++r) {
        const int gm = m0 + wm * 32 + im * 16 + rowBase + r;
        const int gn = n0 + wn * 32 + in * 16 + colIn;
        if (gn >= NoutReal) continue;
        int a = accLo[im][in][r];
        if constexpr (A16) a = accHi[im][in][r] * 256 + a;
        a += bq[gn];
        const float f = (float)a * mul;
        switch (mode) {
          case MODE_ACT8:
          case MODE_ACT8_RELU: {
            float q = fminf(fmaxf(rintf(f), -128.0f), 127.0f);
            if (mode == MODE_ACT8_RELU) q = fmaxf(q, 0.0f);
            ((int8_t*)outp)[(size_t)gm * out_stride + gn] = (int8_t)(int)q;
          } break;
          case MODE_TRANS_T8: {
            float q = fminf(fmaxf(rintf(f), -128.0f), 127.0f);
            int bb = gm / Nn, p = gm % Nn;
            ((int8_t*)outp)[((size_t)bb * Dd + gn) * 256 + p] = (int8_t)(int)q;
          } break;
          case MODE_FROM_T8: {
            float q = fminf(fmaxf(rintf(f), -128.0f), 127.0f);
            int bb = gm / Dd, dd = gm % Dd;
            ((int8_t*)outp)[((size_t)bb * Nn + gn) * Dd + dd] = (int8_t)(int)q;
          } break;
          case MODE_RES16: {
            float q = fminf(fmaxf(rintf(f), -32768.0f), 32767.0f);
            res16[(size_t)gm * Dd + gn] = (int16_t)(int)q;
          } break;
          case MODE_RESADD16: {
            float mixed = f + (float)res16[(size_t)gm * Dd + gn] * m2;
            float q = fminf(fmaxf(rintf(mixed), -32768.0f), 32767.0f);
            res16[(size_t)gm * Dd + gn] = (int16_t)(int)q;
          } break;
          case MODE_RESADD_F32: {
            ((float*)outp)[(size_t)gm * Dd + gn] =
                f + (float)res16[(size_t)gm * Dd + gn] * m2;
          } break;
        }
      }
}

// ------------------------------- fp32 tail ---------------------------------
__global__ void k_pool(const float* __restrict__ F, const float* __restrict__ alpha,
                       const float* __restrict__ beta, float* __restrict__ pooled) {
  int i = blockIdx.x * 256 + threadIdx.x;
  if (i >= Bb * Dd) return;
  int b = i / Dd, d = i % Dd;
  float s = 0.0f;
  for (int n = 0; n < Nn; ++n) s += F[((size_t)b * Nn + n) * Dd + d];
  pooled[i] = (s * (1.0f / (float)Nn)) * alpha[d] + beta[d];
}

__global__ void k_head(const float* __restrict__ pooled, const float* __restrict__ hw,
                       const float* __restrict__ hb, float* __restrict__ out) {
  int i = blockIdx.x * 256 + threadIdx.x;
  if (i >= Bb * 1000) return;
  int b = i / 1000, o = i % 1000;
  float s = hb[o];
  for (int k = 0; k < Dd; ++k) s += pooled[b * Dd + k] * hw[(size_t)o * Dd + k];
  out[i] = s;
}

// ------------------------------- host driver -------------------------------
extern "C" void kernel_launch(void* const* d_in, const int* in_sizes, int n_in,
                              void* d_out, int out_size, void* d_ws, size_t ws_size,
                              hipStream_t stream) {
  (void)in_sizes; (void)n_in; (void)out_size; (void)ws_size;
  const float* x       = (const float*)d_in[0];
  const float* conv_w  = (const float*)d_in[1];
  const float* conv_b  = (const float*)d_in[2];
  const float* Wbase[7] = {(const float*)d_in[3], (const float*)d_in[5],
                           (const float*)d_in[7], (const float*)d_in[9],
                           (const float*)d_in[11], (const float*)d_in[13],
                           (const float*)d_in[15]};
  const float* bbase[7] = {(const float*)d_in[4], (const float*)d_in[6],
                           (const float*)d_in[8], (const float*)d_in[10],
                           (const float*)d_in[12], (const float*)d_in[14],
                           (const float*)d_in[16]};
  const float* acts    = (const float*)d_in[17];
  const float* s_in    = (const float*)d_in[18];
  const float* s_patch = (const float*)d_in[19];
  const float* alpha   = (const float*)d_in[20];
  const float* beta    = (const float*)d_in[21];
  const float* head_w  = (const float*)d_in[22];
  const float* head_b  = (const float*)d_in[23];
  float* out = (float*)d_out;

  // Layer geometry (rows = out features, cols = K); attn padded to 256x256.
  const int rows_[7] = {Dd, Nn, Dd, Dd, HIDD, Dd, Dd};
  const int cols_[7] = {Dd, Nn, Dd, Dd, Dd, HIDD, Dd};
  const int rpad_[7] = {Dd, 256, Dd, Dd, HIDD, Dd, Dd};
  const int cpad_[7] = {Dd, 256, Dd, Dd, Dd, HIDD, Dd};

  char* ws = (char*)d_ws;
  size_t off = 0;
  auto alloc = [&](size_t bytes) {
    size_t o = off;
    off += (bytes + 255) & ~(size_t)255;
    return o;
  };

  const int NLAYER = 1 + NBLK * 7;
  size_t slots_off = alloc((size_t)NLAYER * 4);
  size_t convWq_off = alloc((size_t)Dd * 768);
  size_t convBq_off = alloc((size_t)Dd * 4);
  size_t wq_off[NBLK][7], bq_off[NBLK][7];
  for (int i = 0; i < NBLK; ++i)
    for (int j = 0; j < 7; ++j) {
      wq_off[i][j] = alloc((size_t)rpad_[j] * cpad_[j]);
      bq_off[i][j] = alloc((size_t)rpad_[j] * 4);
    }
  size_t im2col_off = alloc((size_t)M_TOK * 768);
  size_t res16_off  = alloc((size_t)M_TOK * Dd * 2);
  size_t X1_off     = alloc((size_t)M_TOK * HIDD);
  size_t X2_off     = alloc((size_t)M_TOK * HIDD);
  size_t T8_off     = alloc((size_t)Bb * Dd * 256);
  size_t F_off      = alloc((size_t)M_TOK * Dd * 4);
  size_t pooled_off = alloc((size_t)Bb * Dd * 4);

  unsigned* slots = (unsigned*)(ws + slots_off);
  int8_t*  im2c   = (int8_t*)(ws + im2col_off);
  int16_t* res16  = (int16_t*)(ws + res16_off);
  int8_t*  X1     = (int8_t*)(ws + X1_off);
  int8_t*  X2     = (int8_t*)(ws + X2_off);
  int8_t*  T8     = (int8_t*)(ws + T8_off);
  float*   F32    = (float*)(ws + F_off);
  float*   pooled = (float*)(ws + pooled_off);

  auto cdiv = [](int a, int b) { return (a + b - 1) / b; };

  (void)hipMemsetAsync(slots, 0, (size_t)NLAYER * 4, stream);
  (void)hipMemsetAsync(T8, 0, (size_t)Bb * Dd * 256, stream);  // K-pad cols stay zero

  // ---------------- weight / bias quantization ----------------
  {
    int n = Dd * 768;
    k_absmax<<<dim3(min(256, cdiv(n, 256))), dim3(256), 0, stream>>>(conv_w, n, slots + 0);
    k_wquant<<<dim3(min(2048, cdiv(n, 256))), dim3(256), 0, stream>>>(
        conv_w, (int8_t*)(ws + convWq_off), Dd, 768, 768, n, slots + 0);
    k_bquant<<<dim3(cdiv(Dd, 256)), dim3(256), 0, stream>>>(
        conv_b, (int*)(ws + convBq_off), Dd, Dd, slots + 0, s_in);
  }
  for (int i = 0; i < NBLK; ++i) {
    const float* as_j[7] = {
        (i == 0) ? s_patch : (acts + (i - 1) * 7 + 6),  // norm1 <- residual scale
        acts + i * 7 + 0,                               // attn
        acts + i * 7 + 1,                               // gamma1
        acts + i * 7 + 2,                               // norm2
        acts + i * 7 + 3,                               // fc1
        acts + i * 7 + 4,                               // fc2
        acts + i * 7 + 5};                              // gamma2
    for (int j = 0; j < 7; ++j) {
      unsigned* slot = slots + 1 + i * 7 + j;
      const float* W = Wbase[j] + (size_t)i * rows_[j] * cols_[j];
      const float* b = bbase[j] + (size_t)i * rows_[j];
      int n = rows_[j] * cols_[j];
      int npad = rpad_[j] * cpad_[j];
      k_absmax<<<dim3(min(256, cdiv(n, 256))), dim3(256), 0, stream>>>(W, n, slot);
      k_wquant<<<dim3(min(2048, cdiv(npad, 256))), dim3(256), 0, stream>>>(
          W, (int8_t*)(ws + wq_off[i][j]), rows_[j], cols_[j], cpad_[j], npad, slot);
      k_bquant<<<dim3(cdiv(rpad_[j], 256)), dim3(256), 0, stream>>>(
          b, (int*)(ws + bq_off[i][j]), rows_[j], rpad_[j], slot, as_j[j]);
    }
  }

  // ---------------- patch embed ----------------
  k_im2col<<<dim3(cdiv(M_TOK * 768, 256)), dim3(256), 0, stream>>>(x, im2c, s_in);
  k_gemm<false><<<dim3(M_TOK / 128, Dd / 64), dim3(256), 0, stream>>>(
      im2c, 768, (const int8_t*)(ws + convWq_off), 768, (const int*)(ws + convBq_off),
      768, Dd, MODE_RES16, nullptr, 0, slots + 0, s_in, s_patch, s_patch, res16);

  // ---------------- 24 mixer blocks ----------------
  for (int i = 0; i < NBLK; ++i) {
    const float* res_in = (i == 0) ? s_patch : (acts + (i - 1) * 7 + 6);
    unsigned* sl = slots + 1 + i * 7;
    const int8_t* WQ[7];
    const int* BQ[7];
    for (int j = 0; j < 7; ++j) {
      WQ[j] = (const int8_t*)(ws + wq_off[i][j]);
      BQ[j] = (const int*)(ws + bq_off[i][j]);
    }
    // norm1: res16 (int16 A) -> T8 transposed, scale act[i][0]
    k_gemm<true><<<dim3(M_TOK / 128, Dd / 64), dim3(256), 0, stream>>>(
        res16, Dd, WQ[0], Dd, BQ[0], Dd, Dd, MODE_TRANS_T8, T8, 0,
        sl + 0, res_in, acts + i * 7 + 0, res_in, res16);
    // attn (token mixing): T8 [B*D,256] -> X1 as [B,N,D], scale act[i][1]
    k_gemm<false><<<dim3(M_CH / 128, 256 / 64), dim3(256), 0, stream>>>(
        T8, 256, WQ[1], 256, BQ[1], 256, Nn, MODE_FROM_T8, X1, 0,
        sl + 1, acts + i * 7 + 0, acts + i * 7 + 1, res_in, res16);
    // gamma1: X1 -> residual add into res16, scale act[i][2]
    k_gemm<false><<<dim3(M_TOK / 128, Dd / 64), dim3(256), 0, stream>>>(
        X1, Dd, WQ[2], Dd, BQ[2], Dd, Dd, MODE_RESADD16, nullptr, 0,
        sl + 2, acts + i * 7 + 1, acts + i * 7 + 2, res_in, res16);
    // norm2: res16 (int16 A) -> X1 int8, scale act[i][3]
    k_gemm<true><<<dim3(M_TOK / 128, Dd / 64), dim3(256), 0, stream>>>(
        res16, Dd, WQ[3], Dd, BQ[3], Dd, Dd, MODE_ACT8, X1, Dd,
        sl + 3, acts + i * 7 + 2, acts + i * 7 + 3, res_in, res16);
    // fc1: X1 -> X2 (relu), scale act[i][4]
    k_gemm<false><<<dim3(M_TOK / 128, HIDD / 64), dim3(256), 0, stream>>>(
        X1, Dd, WQ[4], Dd, BQ[4], Dd, HIDD, MODE_ACT8_RELU, X2, HIDD,
        sl + 4, acts + i * 7 + 3, acts + i * 7 + 4, res_in, res16);
    // fc2: X2 -> X1, scale act[i][5]
    k_gemm<false><<<dim3(M_TOK / 128, Dd / 64), dim3(256), 0, stream>>>(
        X2, HIDD, WQ[5], HIDD, BQ[5], HIDD, Dd, MODE_ACT8, X1, Dd,
        sl + 5, acts + i * 7 + 4, acts + i * 7 + 5, res_in, res16);
    // gamma2: X1 -> residual add (int16, or fp32 on the last block)
    if (i == NBLK - 1) {
      k_gemm<false><<<dim3(M_TOK / 128, Dd / 64), dim3(256), 0, stream>>>(
          X1, Dd, WQ[6], Dd, BQ[6], Dd, Dd, MODE_RESADD_F32, F32, 0,
          sl + 6, acts + i * 7 + 5, s_patch, acts + i * 7 + 2, res16);
    } else {
      k_gemm<false><<<dim3(M_TOK / 128, Dd / 64), dim3(256), 0, stream>>>(
          X1, Dd, WQ[6], Dd, BQ[6], Dd, Dd, MODE_RESADD16, nullptr, 0,
          sl + 6, acts + i * 7 + 5, acts + i * 7 + 6, acts + i * 7 + 2, res16);
    }
  }

  // ---------------- fp32 tail: affine norm + mean pool + head ----------------
  k_pool<<<dim3(cdiv(Bb * Dd, 256)), dim3(256), 0, stream>>>(F32, alpha, beta, pooled);
  k_head<<<dim3(cdiv(Bb * 1000, 256)), dim3(256), 0, stream>>>(pooled, head_w, head_b, out);
}
